// global_conv_attn_34144990003586
// MI455X (gfx1250) — matrix operand order repaired and checked by reading, not compiled, not run on hardware
//
#include <hip/hip_runtime.h>
#include <hip/hip_bf16.h>

// ---------------- types ----------------
typedef __attribute__((ext_vector_type(16))) __bf16 v16bf;
typedef __attribute__((ext_vector_type(8)))  __bf16 v8bf;
typedef __attribute__((ext_vector_type(8)))  float  v8f;
typedef __attribute__((ext_vector_type(4)))  unsigned int u32x4;
typedef __attribute__((ext_vector_type(8)))  int i32x8;
typedef __attribute__((ext_vector_type(4)))  int i32x4;

#define BM 128
#define BN 128
#define BK 32

__device__ __forceinline__ unsigned short f32_to_bf16_rne(float f) {
    unsigned int u = __builtin_bit_cast(unsigned int, f);
    unsigned int r = u + 0x7FFFu + ((u >> 16) & 1u);
    return (unsigned short)(r >> 16);
}

__device__ __forceinline__ v16bf concat8(v8bf lo, v8bf hi) {
    return __builtin_shufflevector(lo, hi, 0, 1, 2, 3, 4, 5, 6, 7,
                                           8, 9, 10, 11, 12, 13, 14, 15);
}

// ---------------- TDM 2D tile load: global (bf16, row-major, ld elems) -> LDS packed ----------------
// Builds D# per ISA 8.3/8.4: group0 {count=1, lds_addr, global_addr, type=2},
// group1 {data_size=2B, tensor_dim0/1, tile_dim0/1, tensor_dim0_stride}.
__device__ __forceinline__ void tdm_load_2d(unsigned lds_byte_addr,
                                            const unsigned short* gptr,
                                            unsigned tile0,   // tile width (K elems)
                                            unsigned tile1,   // tile height (rows)
                                            unsigned ld,      // row stride (elems)
                                            unsigned rows)    // tensor rows (OOB dim)
{
    unsigned long long ga = (unsigned long long)(size_t)gptr;
    u32x4 g0;
    g0[0] = 1u;                                         // count=1, user D#
    g0[1] = lds_byte_addr;                              // LDS byte address
    g0[2] = (unsigned)(ga & 0xFFFFFFFFu);               // global_addr[31:0]
    g0[3] = ((unsigned)(ga >> 32) & 0x01FFFFFFu)        // global_addr[56:32]
            | 0x80000000u;                              // type=2 ("image")
    unsigned dim0 = ld;                                 // tensor_dim0 (row len)
    unsigned dim1 = rows;                               // tensor_dim1
    unsigned long long s0 = (unsigned long long)ld;     // tensor_dim0_stride
    unsigned long long s1 = (unsigned long long)ld * (unsigned long long)rows;
    i32x8 g1;
    g1[0] = 0x00010000;                                 // data_size=1 -> 2 bytes
    g1[1] = (int)((dim0 & 0xFFFFu) << 16);              // tensor_dim0[15:0]
    g1[2] = (int)((dim0 >> 16) | ((dim1 & 0xFFFFu) << 16));
    g1[3] = (int)((dim1 >> 16) | (tile0 << 16));        // tile_dim0
    g1[4] = (int)(tile1 & 0xFFFFu);                     // tile_dim1 (tile_dim2=0)
    g1[5] = (int)(unsigned)(s0 & 0xFFFFFFFFu);
    g1[6] = (int)(((unsigned)(s0 >> 32) & 0xFFFFu) |
                  (((unsigned)(s1 & 0xFFFFu)) << 16));
    g1[7] = (int)((unsigned)((s1 >> 16) & 0xFFFFFFFFu));
    i32x4 z4 = {0, 0, 0, 0};
#if defined(__clang_major__) && (__clang_major__ >= 23)
    i32x8 z8 = {0, 0, 0, 0, 0, 0, 0, 0};
    __builtin_amdgcn_tensor_load_to_lds(g0, g1, z4, z4, z8, 0);
#else
    __builtin_amdgcn_tensor_load_to_lds(g0, g1, z4, z4, 0);
#endif
}

// ---------------- f32 -> bf16 convert ----------------
__global__ void cvt_f32_bf16_k(const float* __restrict__ in,
                               unsigned short* __restrict__ out, size_t n) {
    size_t i = (size_t)blockIdx.x * blockDim.x + threadIdx.x;
    if (i < n) out[i] = f32_to_bf16_rne(in[i]);
}

// ---------------- f32 [C][N] -> bf16 transposed [N][C] (per batch) ----------------
__global__ void cvt_transpose_bf16_k(const float* __restrict__ in,
                                     unsigned short* __restrict__ out,
                                     int C, int N) {
    size_t i = (size_t)blockIdx.x * blockDim.x + threadIdx.x;  // over N*C, c fast
    if (i >= (size_t)N * C) return;
    int c = (int)(i % C);
    int n = (int)(i / C);
    const float* inb = in + (size_t)blockIdx.y * C * N;
    unsigned short* ob = out + (size_t)blockIdx.y * N * C;
    ob[i] = f32_to_bf16_rne(inb[(size_t)c * N + n]);
}

// ---------------- generic WMMA GEMM with TDM double-buffered staging ----------------
// C[m,n] = sum_k A[m*lda + k] * BT[n*ldbt + k]  (+ bias[m]), f32 accumulate.
// Optional outputs: Cf (f32 [M][N]), Cb (bf16 [M][N]), CT (bf16 [N][M], ld=ldct).
// M,N multiples of 128; K multiple of 32.
__global__ __launch_bounds__(256)
void wmma_gemm_tdm_k(const unsigned short* __restrict__ A, long long aBatch, int lda,
                     const unsigned short* __restrict__ BT, long long bBatch, int ldbt,
                     const float* __restrict__ bias,
                     float* __restrict__ Cf,
                     unsigned short* __restrict__ Cb,
                     unsigned short* __restrict__ CT,
                     long long cBatch, int ldc, int ldct,
                     int M, int N, int K)
{
    // TDM writes tiles packed row-major: a[buf][BM][BK], bt[buf][BN][BK]
    __shared__ unsigned short lds_a[2][BM][BK];
    __shared__ unsigned short lds_bt[2][BN][BK];

    const int tid  = threadIdx.x;
    const int lane = tid & 31;
    const int wave = tid >> 5;                 // 8 waves
    const int wm   = (wave & 3) * 32;          // wave M offset
    const int wn   = (wave >> 2) * 64;         // wave N offset
    const int hi   = lane >> 4;
    const int l15  = lane & 15;

    const int bm0 = blockIdx.y * BM;
    const int bn0 = blockIdx.x * BN;
    const size_t z = blockIdx.z;

    A  += z * (size_t)aBatch;
    BT += z * (size_t)bBatch;

    const unsigned a_lds0 = (unsigned)(size_t)(void*)&lds_a[0][0][0];
    const unsigned b_lds0 = (unsigned)(size_t)(void*)&lds_bt[0][0][0];
    const unsigned ABYTES = BM * BK * 2;
    const unsigned BBYTES = BN * BK * 2;

    const v8f vzero = {0.f, 0.f, 0.f, 0.f, 0.f, 0.f, 0.f, 0.f};
    v8f acc[2][4];
    #pragma unroll
    for (int i = 0; i < 2; ++i)
        #pragma unroll
        for (int j = 0; j < 4; ++j) acc[i][j] = vzero;

    // ---- prologue: TDM-stage K-tile 0 into buffer 0 ----
    if (wave == 0) {
        tdm_load_2d(a_lds0, A + (size_t)bm0 * lda, BK, BM, (unsigned)lda, (unsigned)M);
        tdm_load_2d(b_lds0, BT + (size_t)bn0 * ldbt, BK, BN, (unsigned)ldbt, (unsigned)N);
        __builtin_amdgcn_s_wait_tensorcnt(0);
    }
    __syncthreads();

    int cur = 0;
    for (int k0 = 0; k0 < K; k0 += BK) {
        // ---- async TDM fetch of next K-tile into the other buffer ----
        if ((k0 + BK < K) && wave == 0) {
            int nxt = cur ^ 1;
            tdm_load_2d(a_lds0 + nxt * ABYTES,
                        A + (size_t)bm0 * lda + (k0 + BK), BK, BM,
                        (unsigned)lda, (unsigned)M);
            tdm_load_2d(b_lds0 + nxt * BBYTES,
                        BT + (size_t)bn0 * ldbt + (k0 + BK), BK, BN,
                        (unsigned)ldbt, (unsigned)N);
        }

        // ---- fragments: per-lane K-contiguous 16B runs -> ds_load_b128 ----
        // A (16-bit 16x32, ISA 7.12.2): lane group hi: K chunks at hi*8 and hi*8+16
        v16bf afr[2];
        #pragma unroll
        for (int i = 0; i < 2; ++i) {
            const unsigned short* rp = &lds_a[cur][wm + i * 16 + l15][0];
            v8bf alo = *(const v8bf*)(rp + hi * 8);
            v8bf ahi = *(const v8bf*)(rp + hi * 8 + 16);
            afr[i] = concat8(alo, ahi);
        }
        // B (32x16): lane group hi: K chunks at hi*8 and hi*8+16 of column l15
        v16bf bfr[4];
        #pragma unroll
        for (int j4 = 0; j4 < 4; ++j4) {
            const unsigned short* cp = &lds_bt[cur][wn + j4 * 16 + l15][0];
            v8bf blo = *(const v8bf*)(cp + hi * 8);
            v8bf bhi = *(const v8bf*)(cp + hi * 8 + 16);
            bfr[j4] = concat8(blo, bhi);
        }

        // ---- 8 WMMAs per wave per K-step ----
        #pragma unroll
        for (int i = 0; i < 2; ++i)
            #pragma unroll
            for (int j = 0; j < 4; ++j)
                acc[i][j] = __builtin_amdgcn_wmma_f32_16x16x32_bf16(
                    false, afr[i], false, bfr[j], (short)0, acc[i][j],
                    false, false);

        if (wave == 0) __builtin_amdgcn_s_wait_tensorcnt(0);
        __syncthreads();
        cur ^= 1;
    }

    // ---- epilogue: C/D layout: VGPR r -> M = r + 8*hi, N = lane&15 ----
    #pragma unroll
    for (int i = 0; i < 2; ++i) {
        int mbase = bm0 + wm + i * 16 + 8 * hi;
        #pragma unroll
        for (int j = 0; j < 4; ++j) {
            int ncol = bn0 + wn + j * 16 + l15;
            #pragma unroll
            for (int r = 0; r < 8; ++r) {
                int m = mbase + r;
                float val = acc[i][j][r];
                if (bias) val += bias[m];
                if (Cf) Cf[z * (size_t)cBatch + (size_t)m * ldc + ncol] = val;
                if (Cb) Cb[z * (size_t)cBatch + (size_t)m * ldc + ncol] =
                        f32_to_bf16_rne(val);
                if (CT) CT[z * (size_t)cBatch + (size_t)ncol * ldct + m] =
                        f32_to_bf16_rne(val);
            }
        }
    }
}

// ---------------- grouped 3-tap conv + ReLU, writes bf16 transposed [N][H] ----------------
__global__ void grouped_conv3_relu_k(const float* __restrict__ v1,
                                     const float* __restrict__ W2,
                                     const float* __restrict__ b2,
                                     unsigned short* __restrict__ v2T,
                                     int Bc, int H, int N)
{
    size_t idx = (size_t)blockIdx.x * blockDim.x + threadIdx.x;  // h fast
    size_t total = (size_t)Bc * H * N;
    if (idx >= total) return;
    int h = (int)(idx % H);
    int n = (int)((idx / H) % N);
    int b = (int)(idx / ((size_t)N * H));
    int g = h >> 3;  // 8 channels per group
    float acc = b2[h];
    const float* vb = v1 + ((size_t)b * H + (size_t)g * 8) * N;
    const float* w  = W2 + (size_t)h * 24;
    #pragma unroll
    for (int i = 0; i < 8; ++i) {
        const float* row = vb + (size_t)i * N;
        if (n > 0)     acc += w[i * 3 + 0] * row[n - 1];
                       acc += w[i * 3 + 1] * row[n];
        if (n < N - 1) acc += w[i * 3 + 2] * row[n + 1];
    }
    acc = acc > 0.f ? acc : 0.f;
    v2T[idx] = f32_to_bf16_rne(acc);   // v2T[b][n][h]
}

// ---------------- column softmax over n (axis=-2), writes P^T[m][n] bf16 ----------------
__global__ __launch_bounds__(256)
void softmax_cols_k(const float* __restrict__ S,
                    unsigned short* __restrict__ PT, int N)
{
    int m = blockIdx.x * 256 + threadIdx.x;
    const float* Sb = S + (size_t)blockIdx.y * N * N;
    unsigned short* Pb = PT + (size_t)blockIdx.y * N * N + (size_t)m * N;
    float maxv = -3.4e38f, sum = 0.f;
    for (int n = 0; n < N; ++n) {
        float s = Sb[(size_t)n * N + m];
        if (s > maxv) { sum *= __expf(maxv - s); maxv = s; }
        sum += __expf(s - maxv);
    }
    float inv = 1.0f / sum;
    for (int n = 0; n < N; ++n) {
        float s = Sb[(size_t)n * N + m];
        Pb[n] = f32_to_bf16_rne(__expf(s - maxv) * inv);  // PT[m][n]
    }
}

// ---------------- host-side orchestration ----------------
extern "C" void kernel_launch(void* const* d_in, const int* in_sizes, int n_in,
                              void* d_out, int out_size, void* d_ws, size_t ws_size,
                              hipStream_t stream) {
    const float* x  = (const float*)d_in[0];
    const float* Wk = (const float*)d_in[1];
    const float* bk = (const float*)d_in[2];
    const float* Wq = (const float*)d_in[3];
    const float* bq = (const float*)d_in[4];
    const float* W1 = (const float*)d_in[5];
    const float* b1 = (const float*)d_in[6];
    const float* W2 = (const float*)d_in[7];
    const float* b2 = (const float*)d_in[8];
    const float* W3 = (const float*)d_in[9];
    const float* b3 = (const float*)d_in[10];
    float* out = (float*)d_out;

    const int Bc = 4, C = 256, H = 256, O = 256, N = 4096;
    const size_t XE = (size_t)Bc * C * N;
    const size_t HE = (size_t)Bc * H * N;
    const size_t SE = (size_t)Bc * N * N;
    const size_t WE = (size_t)H * C;

    char* ws = (char*)d_ws;
    size_t off = 0;
    auto alloc = [&](size_t bytes) -> void* {
        void* p = ws + off;
        off = (off + bytes + 255) & ~(size_t)255;
        return p;
    };
    unsigned short* xbT = (unsigned short*)alloc(XE * 2);  // [B][N][C]
    unsigned short* wkb = (unsigned short*)alloc(WE * 2);
    unsigned short* wqb = (unsigned short*)alloc(WE * 2);
    unsigned short* w1b = (unsigned short*)alloc(WE * 2);
    unsigned short* w3b = (unsigned short*)alloc(WE * 2);
    unsigned short* kT  = (unsigned short*)alloc(HE * 2);  // [B][N][H]
    unsigned short* qT  = (unsigned short*)alloc(HE * 2);  // [B][N][H]
    float*          v1f = (float*)alloc(HE * 4);           // [B][H][N]
    unsigned short* v2T = (unsigned short*)alloc(HE * 2);  // [B][N][H]
    unsigned short* vb  = (unsigned short*)alloc(HE * 2);  // [B][O][N]
    float*          Sbf = (float*)alloc(SE * 4);           // [B][N][N]
    unsigned short* PT  = (unsigned short*)alloc(SE * 2);  // [B][N(m)][N(n)]

    // --- bf16 conversions ---
    dim3 gXT((unsigned)(((size_t)N * C + 255) / 256), Bc);
    cvt_transpose_bf16_k<<<gXT, 256, 0, stream>>>(x, xbT, C, N);
    cvt_f32_bf16_k<<<(unsigned)((WE + 255) / 256), 256, 0, stream>>>(Wk, wkb, WE);
    cvt_f32_bf16_k<<<(unsigned)((WE + 255) / 256), 256, 0, stream>>>(Wq, wqb, WE);
    cvt_f32_bf16_k<<<(unsigned)((WE + 255) / 256), 256, 0, stream>>>(W1, w1b, WE);
    cvt_f32_bf16_k<<<(unsigned)((WE + 255) / 256), 256, 0, stream>>>(W3, w3b, WE);

    // --- k, q (bf16 transposed out) and v1 (f32 out): [H,C] @ [C,N] per batch ---
    dim3 gHC(N / BN, H / BM, Bc);
    wmma_gemm_tdm_k<<<gHC, 256, 0, stream>>>(
        wkb, 0, C, xbT, (long long)N * C, C, bk,
        nullptr, nullptr, kT, (long long)N * H, N, H, H, N, C);
    wmma_gemm_tdm_k<<<gHC, 256, 0, stream>>>(
        wqb, 0, C, xbT, (long long)N * C, C, bq,
        nullptr, nullptr, qT, (long long)N * H, N, H, H, N, C);
    wmma_gemm_tdm_k<<<gHC, 256, 0, stream>>>(
        w1b, 0, C, xbT, (long long)N * C, C, b1,
        v1f, nullptr, nullptr, (long long)H * N, N, H, H, N, C);

    // --- grouped conv + ReLU -> v2^T bf16 ---
    grouped_conv3_relu_k<<<(unsigned)((HE + 255) / 256), 256, 0, stream>>>(
        v1f, W2, b2, v2T, Bc, H, N);

    // --- v = W3 @ v2 + b3 -> bf16 [O][N] ---
    wmma_gemm_tdm_k<<<gHC, 256, 0, stream>>>(
        w3b, 0, H, v2T, (long long)N * H, H, b3,
        nullptr, vb, nullptr, (long long)O * N, N, H, O, N, H);

    // --- S = k^T q : A = kT [N][H], BT = qT [N][H] ---
    dim3 gS(N / BN, N / BM, Bc);
    wmma_gemm_tdm_k<<<gS, 256, 0, stream>>>(
        kT, (long long)N * H, H, qT, (long long)N * H, H, nullptr,
        Sbf, nullptr, nullptr, (long long)N * N, N, N, N, N, H);

    // --- P^T = softmax over rows (axis=-2) ---
    dim3 gSm(N / 256, Bc);
    softmax_cols_k<<<gSm, 256, 0, stream>>>(Sbf, PT, N);

    // --- out = v @ P : A = vb [O][N], BT = PT [N(m)][N(n)] , K = N ---
    dim3 gO(N / BN, O / BM, Bc);
    wmma_gemm_tdm_k<<<gO, 256, 0, stream>>>(
        vb, (long long)O * N, N, PT, (long long)N * N, N, nullptr,
        out, nullptr, nullptr, (long long)O * N, N, O, O, N, N);
}
